// SpeakingEncoder_40003325395700
// MI455X (gfx1250) — matrix operand order repaired and checked by reading, not compiled, not run on hardware
//
#include <hip/hip_runtime.h>

// ---------------------------------------------------------------------------
// Problem constants (from reference): B=8, T=2048, D=1024, K=4096
// ---------------------------------------------------------------------------
#define T_SEQ   2048
#define DIM     1024
#define KCODES  4096
#define M_TOTAL (8 * 2048)          // B*T = 16384 rows

#define M_TILE  128                 // rows per workgroup
#define N_TILE  128                 // codes per inner tile
#define N_CHUNK 1024                // codes swept per workgroup (4 chunks total)
#define KC      64                  // D-chunk staged per iteration (2 wmma k-steps)
#define NT_PER_WG (N_CHUNK / N_TILE)        // 8
#define DK_ITERS  (DIM / KC)                // 16
#define TILES_TOTAL (NT_PER_WG * DK_ITERS)  // 128 staged B tiles per WG

#define A_STRIDE 1048               // 1024 + 24 pad (bf16); 2096B row, 16B aligned, conflict-free
#define B_STRIDE 72                 // 64 + 8 pad   (bf16); 144B row, 16B aligned, conflict-free
#define A_BYTES (M_TILE * A_STRIDE * 2)          // 268288
#define B_ELEMS (N_TILE * B_STRIDE)              // bf16 elems per B buffer
#define B_BYTES (B_ELEMS * 2)                    // 18432
#define SMEM_BYTES (A_BYTES + 2 * B_BYTES + M_TILE * 8)   // 306176 <= 320K

typedef __attribute__((ext_vector_type(16))) __bf16 v16bf;
typedef __attribute__((ext_vector_type(8)))  __bf16 v8bf;
typedef __attribute__((ext_vector_type(8)))  float  v8f;

union FragU {
    v16bf v;
    struct { v8bf lo; v8bf hi; } p;
};

// ---------------------------------------------------------------------------
// Kernel 1: c2[k] = sum_d codebook[k][d]^2   (one wave per code row)
// ---------------------------------------------------------------------------
__global__ void vq_c2_kernel(const float* __restrict__ cb, float* __restrict__ c2) {
    const int gid  = blockIdx.x * blockDim.x + threadIdx.x;
    const int row  = gid >> 5;
    const int lane = gid & 31;
    const float4* p = (const float4*)(cb + (size_t)row * DIM);
    float s = 0.f;
    #pragma unroll 4
    for (int j = lane; j < DIM / 4; j += 32) {
        float4 v = p[j];
        s += v.x * v.x + v.y * v.y + v.z * v.z + v.w * v.w;
    }
    #pragma unroll
    for (int m = 16; m; m >>= 1) s += __shfl_xor(s, m, 32);
    if (lane == 0) c2[row] = s;
}

// ---------------------------------------------------------------------------
// Kernel 2: fused bf16-WMMA GEMM + argmin over a 1024-code N-chunk.
// grid = (M_TOTAL/M_TILE, KCODES/N_CHUNK) = 128 x 4 = 512 workgroups.
// 256 threads = 8 waves, wave grid 4(M) x 2(N); each wave owns a 32x64 C tile.
// B tiles are software-pipelined through a ping-pong LDS buffer: next tile's
// global loads are in flight while WMMAs consume the current tile.
// ---------------------------------------------------------------------------
__global__ __launch_bounds__(256, 1)
void vq_argmin_kernel(const float* __restrict__ x, const float* __restrict__ cb,
                      const float* __restrict__ c2,
                      unsigned long long* __restrict__ gbest) {
    extern __shared__ char smem[];
    __bf16 (*As)[A_STRIDE] = (__bf16 (*)[A_STRIDE])(smem);
    __bf16* Bsb = (__bf16*)(smem + A_BYTES);                 // 2 x B_ELEMS ping-pong
    unsigned long long* best = (unsigned long long*)(smem + A_BYTES + 2 * B_BYTES);

    const int tid    = threadIdx.x;
    const int wave   = tid >> 5;
    const int lane   = tid & 31;
    const int half   = lane >> 4;       // 0: lanes 0-15, 1: lanes 16-31
    const int l15    = lane & 15;
    const int wave_m = wave >> 1;       // 0..3 -> 32-row strip
    const int wave_n = wave & 1;        // 0..1 -> 64-col strip
    const int mbase  = blockIdx.x * M_TILE;
    const int nchunk = blockIdx.y * N_CHUNK;

    // B staging assignment: 2 threads per codebook row, 32 floats each
    const int brow = tid >> 1;
    const int bc0  = (tid & 1) * 32;

    if (tid < M_TILE) best[tid] = ~0ull;

    // Stage the x slab once, f32 -> bf16, into persistent LDS (262 KB).
    for (int lm = 0; lm < M_TILE; ++lm) {
        float4 v = *(const float4*)(x + (size_t)(mbase + lm) * DIM + tid * 4);
        __bf16* d = &As[lm][tid * 4];
        d[0] = (__bf16)v.x; d[1] = (__bf16)v.y; d[2] = (__bf16)v.z; d[3] = (__bf16)v.w;
    }

    // Pipeline prologue: fetch B tile 0 into registers, store into buffer 0.
    float4 breg[8];
    {
        const float4* src = (const float4*)(cb + (size_t)(nchunk + brow) * DIM + bc0);
        #pragma unroll
        for (int j = 0; j < 8; ++j) breg[j] = src[j];
        __bf16* d = Bsb + brow * B_STRIDE + bc0;
        #pragma unroll
        for (int j = 0; j < 8; ++j) {
            d[j * 4 + 0] = (__bf16)breg[j].x; d[j * 4 + 1] = (__bf16)breg[j].y;
            d[j * 4 + 2] = (__bf16)breg[j].z; d[j * 4 + 3] = (__bf16)breg[j].w;
        }
    }
    __syncthreads();

    const v8f vzero = {0.f, 0.f, 0.f, 0.f, 0.f, 0.f, 0.f, 0.f};
    int cur = 0;

    for (int nt = 0; nt < NT_PER_WG; ++nt) {
        const int nbase = nchunk + nt * N_TILE;

        v8f acc[2][4];
        #pragma unroll
        for (int tm = 0; tm < 2; ++tm)
            #pragma unroll
            for (int tn = 0; tn < 4; ++tn) acc[tm][tn] = vzero;

        for (int dki = 0; dki < DK_ITERS; ++dki) {
            const int dk = dki * KC;

            // ---- issue next tile's global loads (overlap with WMMA below) ----
            const int itn = nt * DK_ITERS + dki + 1;       // linear successor tile
            const bool have_next = itn < TILES_TOTAL;      // uniform across block
            if (have_next) {
                const int nnb = nchunk + (itn >> 4) * N_TILE;   // successor nbase
                const int ndk = (itn & (DK_ITERS - 1)) * KC;    // successor dk
                const float4* src =
                    (const float4*)(cb + (size_t)(nnb + brow) * DIM + ndk + bc0);
                #pragma unroll
                for (int j = 0; j < 8; ++j) breg[j] = src[j];
            }

            // ---- WMMA on the current LDS buffer ----
            const __bf16* Bcur = Bsb + cur * B_ELEMS;
            #pragma unroll
            for (int ks = 0; ks < 2; ++ks) {
                const int kk = dk + ks * 32;   // global k of this wmma step
                FragU a[2], b[4];
                // A frag (16x32 bf16): lanes<16 hold K {0..7,16..23}, lanes>=16 {8..15,24..31}
                #pragma unroll
                for (int tm = 0; tm < 2; ++tm) {
                    const __bf16* r = As[wave_m * 32 + tm * 16 + l15];
                    a[tm].p.lo = *(const v8bf*)(r + kk + 8 * half);
                    a[tm].p.hi = *(const v8bf*)(r + kk + 16 + 8 * half);
                }
                // B frag (32x16 bf16): lanes<16 hold K {0..7,16..23}, lanes>=16 {8..15,24..31}
                #pragma unroll
                for (int tn = 0; tn < 4; ++tn) {
                    const __bf16* r = Bcur + (wave_n * 64 + tn * 16 + l15) * B_STRIDE;
                    const int kl = ks * 32 + 8 * half;
                    b[tn].p.lo = *(const v8bf*)(r + kl);
                    b[tn].p.hi = *(const v8bf*)(r + kl + 16);
                }
                #pragma unroll
                for (int tm = 0; tm < 2; ++tm)
                    #pragma unroll
                    for (int tn = 0; tn < 4; ++tn)
                        acc[tm][tn] = __builtin_amdgcn_wmma_f32_16x16x32_bf16(
                            false, a[tm].v, false, b[tn].v,
                            (short)0, acc[tm][tn], false, false);
            }

            // ---- convert + store next tile into the other buffer ----
            if (have_next) {
                __bf16* d = Bsb + (cur ^ 1) * B_ELEMS + brow * B_STRIDE + bc0;
                #pragma unroll
                for (int j = 0; j < 8; ++j) {
                    d[j * 4 + 0] = (__bf16)breg[j].x; d[j * 4 + 1] = (__bf16)breg[j].y;
                    d[j * 4 + 2] = (__bf16)breg[j].z; d[j * 4 + 3] = (__bf16)breg[j].w;
                }
            }
            __syncthreads();     // one barrier per pipelined iteration
            cur ^= 1;
        }

        // Epilogue: d2 = c2[n] - 2*s ; fold into per-row best (key = ordkey<<32 | n)
        #pragma unroll
        for (int tm = 0; tm < 2; ++tm) {
            #pragma unroll
            for (int r = 0; r < 8; ++r) {
                const int mloc = wave_m * 32 + tm * 16 + half * 8 + r;
                unsigned long long bk = ~0ull;
                #pragma unroll
                for (int tn = 0; tn < 4; ++tn) {
                    const int n = nbase + wave_n * 64 + tn * 16 + l15;
                    const float d2 = c2[n] - 2.0f * acc[tm][tn][r];
                    unsigned u = __float_as_uint(d2);
                    u = (u & 0x80000000u) ? ~u : (u | 0x80000000u); // order-preserving
                    const unsigned long long key =
                        ((unsigned long long)u << 32) | (unsigned long long)(unsigned)n;
                    bk = key < bk ? key : bk;
                }
                // reduce over the 16 N-lanes of this half-wave
                #pragma unroll
                for (int mm = 1; mm < 16; mm <<= 1) {
                    const unsigned long long o = __shfl_xor(bk, mm, 32);
                    bk = o < bk ? o : bk;
                }
                if (l15 == 0) atomicMin(&best[mloc], bk);   // ds_min_u64
            }
        }
    }
    __syncthreads();

    // Publish per-row best across the 4 N-chunks.
    if (tid < M_TILE) atomicMin(&gbest[mbase + tid], best[tid]);  // global_atomic_min_u64
}

// ---------------------------------------------------------------------------
// Kernel 3: out[m] = embed_table[idx[m]] + pe(t) ; 8 rows per block.
// ---------------------------------------------------------------------------
__global__ __launch_bounds__(256)
void vq_gather_pe_kernel(const float* __restrict__ et,
                         const unsigned long long* __restrict__ gbest,
                         float* __restrict__ out) {
    const int tid = threadIdx.x;
    const float C  = -9.210340371976184f / (float)DIM;    // -ln(10000)/D
    const float f0 = __expf((float)(tid * 4) * C);        // div_term for pair i = 2*tid
    const float f1 = __expf((float)(tid * 4 + 2) * C);    // div_term for pair i = 2*tid+1

    #pragma unroll
    for (int j = 0; j < 8; ++j) {
        const size_t m   = (size_t)blockIdx.x * 8 + j;
        const unsigned idx = (unsigned)gbest[m];          // low 32 bits = code index
        const float pos  = (float)(int)(m & (T_SEQ - 1)); // t = m % T
        float4 e = *(const float4*)(et + (size_t)idx * DIM + tid * 4);
        const float a0 = pos * f0, a1 = pos * f1;
        float4 o;
        o.x = e.x + __sinf(a0);
        o.y = e.y + __cosf(a0);
        o.z = e.z + __sinf(a1);
        o.w = e.w + __cosf(a1);
        *(float4*)(out + m * DIM + tid * 4) = o;
    }
}

// ---------------------------------------------------------------------------
// Host launch
// ---------------------------------------------------------------------------
extern "C" void kernel_launch(void* const* d_in, const int* in_sizes, int n_in,
                              void* d_out, int out_size, void* d_ws, size_t ws_size,
                              hipStream_t stream) {
    (void)in_sizes; (void)n_in; (void)out_size; (void)ws_size;

    const float* x  = (const float*)d_in[0];   // [B,T,D] f32
    const float* cb = (const float*)d_in[1];   // [K,D]  f32
    const float* et = (const float*)d_in[2];   // [K,D]  f32
    float* out = (float*)d_out;                // [B,T,D] f32

    // Workspace layout: [0,16KB) c2 ; [16KB, 16KB+128KB) per-row best (u64)
    float* c2 = (float*)d_ws;
    unsigned long long* gbest = (unsigned long long*)((char*)d_ws + KCODES * sizeof(float));

    // 1) per-code squared norms (one wave per code)
    vq_c2_kernel<<<KCODES * 32 / 256, 256, 0, stream>>>(cb, c2);

    // 2) init global best keys to +inf (0xFF fill == ~0ull)
    hipMemsetAsync(gbest, 0xFF, (size_t)M_TOTAL * sizeof(unsigned long long), stream);

    // 3) fused bf16 WMMA GEMM + argmin (software-pipelined B staging)
    static_assert(SMEM_BYTES <= 320 * 1024, "LDS budget exceeded");
    hipFuncSetAttribute((const void*)vq_argmin_kernel,
                        hipFuncAttributeMaxDynamicSharedMemorySize, SMEM_BYTES);
    dim3 grid(M_TOTAL / M_TILE, KCODES / N_CHUNK);   // 128 x 4 = 512 workgroups
    vq_argmin_kernel<<<grid, 256, SMEM_BYTES, stream>>>(x, cb, c2, gbest);

    // 4) embedding gather + positional encoding
    vq_gather_pe_kernel<<<M_TOTAL / 8, 256, 0, stream>>>(et, gbest, out);
}
